// FiLM_84335977824271
// MI455X (gfx1250) — compile-verified, matrix operand order checked by reading, not run on hardware
//
#include <hip/hip_runtime.h>
#include <hip/hip_bf16.h>

// ---------------------------------------------------------------------------
// FiLM forward for MI455X (gfx1250), fully collapsed to three bf16 WMMA GEMMs
// per window:
//   R(1024x8192) = U(1024xws) @ M(wsx8192)       (scan+FFT folded into M)
//   Y(1024x256)  = R @ Wbig(8192x256)            (mode mixing, W input)
//   out(1024xws) = Y @ E^T                       (Legendre decode, scattered)
// Constants (A_d via Gauss-Jordan inverse, K scan, G, M, E) are rebuilt on
// device every call -> deterministic, graph-capture safe.
// Wave tile is 32x64 (2 A-frags x 4 B-frags = 8 WMMA per K-step vs 12 b128
// loads) to raise operand-arithmetic intensity over the 32x32 variant.
// Workspace requirement: ~29 MB.
// ---------------------------------------------------------------------------

typedef __attribute__((ext_vector_type(16))) __bf16 bf16x16;
typedef __attribute__((ext_vector_type(8)))  __bf16 bf16x8;
typedef __attribute__((ext_vector_type(8)))  float  v8f;

#define NROWS     1024            // 8*8*16 flattened (b,h,c)
#define PRED_TOT  672
#define FEL_SCALE (1.0f / (256.0f * 256.0f))

// ---- closed-form LMU transition entries -----------------------------------
__device__ __forceinline__ float Acont(int i, int j) {
  float R = 2.0f * (float)i + 1.0f;
  if (i < j) return -R;
  return ((i - j + 1) & 1) ? -R : R;      // (-1)^(i-j+1) * (2i+1)
}
__device__ __forceinline__ float Bcont(int i) {
  float R = 2.0f * (float)i + 1.0f;
  return (i & 1) ? -R : R;                // (-1)^i * (2i+1)
}

// ---- AUG = [ I - dt/2*A | I ]  (256 x 512) --------------------------------
__global__ void k_build_aug(float* AUG, float dt) {
  int idx = blockIdx.x * blockDim.x + threadIdx.x;
  if (idx >= 256 * 512) return;
  int i = idx >> 9, j = idx & 511;
  float v;
  if (j < 256) v = (i == j ? 1.0f : 0.0f) - 0.5f * dt * Acont(i, j);
  else         v = (i == (j - 256)) ? 1.0f : 0.0f;
  AUG[idx] = v;
}

// ---- Gauss-Jordan: right half of AUG becomes (I - dt/2*A)^-1 --------------
__global__ void k_gauss_jordan(float* AUG) {
  int t = threadIdx.x;                        // 256 threads, 1 block
  for (int p = 0; p < 256; ++p) {
    float piv = AUG[p * 512 + p];
    __syncthreads();
    AUG[p * 512 + t]       /= piv;            // normalize pivot row
    AUG[p * 512 + t + 256] /= piv;
    __syncthreads();
    if (t != p) {                             // eliminate own row
      float f = AUG[t * 512 + p];
      if (f != 0.0f)
        for (int j = 0; j < 512; ++j)
          AUG[t * 512 + j] -= f * AUG[p * 512 + j];
    }
    __syncthreads();
  }
}

// ---- Ad = Minv @ (I + dt/2*A) ; Bd = dt * Minv @ B ------------------------
__global__ void k_compute_Ad(const float* AUG, float* Ad, float dt) {
  int idx = blockIdx.x * blockDim.x + threadIdx.x;
  if (idx >= 256 * 256) return;
  int i = idx >> 8, j = idx & 255;
  const float* Minv = AUG + i * 512 + 256;
  float acc = 0.f;
  for (int m = 0; m < 256; ++m)
    acc += Minv[m] * ((m == j ? 1.0f : 0.0f) + 0.5f * dt * Acont(m, j));
  Ad[idx] = acc;
}
__global__ void k_compute_Bd(const float* AUG, float* Bd, float dt) {
  int i = threadIdx.x;
  const float* Minv = AUG + i * 512 + 256;
  float acc = 0.f;
  for (int m = 0; m < 256; ++m) acc += Minv[m] * Bcont(m);
  Bd[i] = acc * dt;
}

// ---- K[j] = Bd @ Ad^j, row scan kept in LDS -------------------------------
__global__ void k_scan_K(const float* Ad, const float* Bd, float* Kmat, int ws) {
  __shared__ float cur[256];
  int n = threadIdx.x;
  float v = Bd[n];
  cur[n] = v;
  Kmat[n] = v;
  __syncthreads();
  for (int j = 1; j < ws; ++j) {
    float acc = 0.f;
    for (int m = 0; m < 256; ++m) acc += cur[m] * Ad[m * 256 + n];
    Kmat[j * 256 + n] = acc;
    __syncthreads();
    cur[n] = acc;
    __syncthreads();
  }
}

// ---- G[t,k]: single-sample irfft weights, FEL_SCALE folded in -------------
__global__ void k_gen_G(float* G, int ws) {
  int idx = blockIdx.x * blockDim.x + threadIdx.x;
  if (idx >= ws * 32) return;
  int t = idx >> 5, k = idx & 31;
  float wk = (k == 0) ? 1.0f : 2.0f;
  float ph = 2.0f * 3.14159265358979323846f * (float)k *
             (float)(t - (ws - 1)) / (float)ws;
  G[idx] = FEL_SCALE * (wk / (float)ws) * __cosf(ph);
}

// ---- MT[col= k*256+n][s] = sum_j K[j][n]*G[j+s][k]  (B^T of GEMM1, bf16) --
__global__ void k_gen_MT(const float* Kmat, const float* G, __bf16* MT, int ws) {
  int idx = blockIdx.x * blockDim.x + threadIdx.x;
  if (idx >= 8192 * ws) return;
  int s = idx % ws;
  int col = idx / ws;
  int n = col & 255, k = col >> 8;
  float acc = 0.f;
  for (int j = 0; j < ws - s; ++j)
    acc += Kmat[j * 256 + n] * G[(j + s) * 32 + k];
  MT[(size_t)col * ws + s] = (__bf16)acc;
}

// ---- E[p][o] Legendre eval (rows p>=ws zero-padded to npad) ---------------
__global__ void k_gen_E(__bf16* E, int ws, int npad) {
  int p = blockIdx.x * blockDim.x + threadIdx.x;
  if (p >= npad) return;
  __bf16* row = E + (size_t)p * 256;
  if (p >= ws) { for (int n = 0; n < 256; ++n) row[n] = (__bf16)0.0f; return; }
  float x = 1.0f - 2.0f * (float)p / (float)ws;
  float p0 = 1.0f, p1 = x;
  row[0] = (__bf16)p0;
  row[1] = (__bf16)p1;
  for (int n = 1; n < 255; ++n) {
    float p2 = ((2.0f * n + 1.0f) * x * p1 - (float)n * p0) / ((float)n + 1.0f);
    row[n + 1] = (__bf16)p2;
    p0 = p1; p1 = p2;
  }
}

// ---- U[row= (b,h,c)][s] = tensor[b,h,384-ws+s,c] (bf16) -------------------
__global__ void k_gen_U(const float* tensor, __bf16* U, int ws) {
  int idx = blockIdx.x * blockDim.x + threadIdx.x;
  if (idx >= NROWS * ws) return;
  int s = idx % ws, row = idx / ws;
  int bh = row >> 4, c = row & 15;
  int t = 384 - ws + s;
  U[(size_t)row * ws + s] = (__bf16)tensor[bh * 6144 + t * 16 + c];
}

// ---- Wt[o][kb*256+n] = W[n,o,kb]  (B^T of GEMM2, bf16) --------------------
__global__ void k_gen_Wt(const float* W, __bf16* Wt) {
  int idx = blockIdx.x * blockDim.x + threadIdx.x;
  if (idx >= 256 * 8192) return;
  int col = idx & 8191;
  int o = idx >> 13;
  int n = col & 255, kb = col >> 8;
  Wt[(size_t)o * 8192 + col] = (__bf16)W[((size_t)n * 256 + o) * 32 + kb];
}

// ---- WMMA fragment load: A row-major, B pre-transposed --------------------
// 16-bit A 16x32 ISA layout: lane<16 holds K[0..7],[16..23]; lane>=16 holds
// K[8..15],[24..31]; this helper loads both 16B halves as b128.
__device__ __forceinline__ bf16x16 ldfrag(const __bf16* p) {
  bf16x8 lo = *reinterpret_cast<const bf16x8*>(p);
  bf16x8 hi = *reinterpret_cast<const bf16x8*>(p + 16);
  bf16x16 r;
#pragma unroll
  for (int i = 0; i < 8; ++i) { r[i] = lo[i]; r[i + 8] = hi[i]; }
  return r;
}

// ---- tiled bf16 WMMA GEMM: C = A @ B, BT is B transposed ------------------
// 256 threads = 8 waves (4 M x 2 N), 32x64 tile per wave (2x4 WMMA frags),
// block tile 128x128.  8 WMMA : 12 b128 loads per K-step.
// mode 0: store bf16 row-major to outB.  mode 1: scatter f32 into d_out at
// out[(row>>4)*10752 + (p_goff+col)*16 + (row&15)], guarded by col<n_valid.
__global__ void __launch_bounds__(256)
k_wmma_gemm(const __bf16* A, int lda, const __bf16* BT, int ldb, int K,
            int mode, __bf16* outB, int ldc, float* outF,
            int n_valid, int p_goff) {
  const int lane = threadIdx.x & 31;
  const int wave = threadIdx.x >> 5;
  const int wm = wave & 3;                  // 4 waves over M
  const int wn = wave >> 2;                 // 2 waves over N
  const int m0 = blockIdx.y * 128 + wm * 32;
  const int n0 = blockIdx.x * 128 + wn * 64;
  const int r16 = lane & 15;
  const int koff = (lane >> 4) << 3;

  const __bf16* pa0 = A  + (size_t)(m0 + r16)      * lda + koff;
  const __bf16* pa1 = A  + (size_t)(m0 + 16 + r16) * lda + koff;
  const __bf16* pb0 = BT + (size_t)(n0 + r16)      * ldb + koff;
  const __bf16* pb1 = BT + (size_t)(n0 + 16 + r16) * ldb + koff;
  const __bf16* pb2 = BT + (size_t)(n0 + 32 + r16) * ldb + koff;
  const __bf16* pb3 = BT + (size_t)(n0 + 48 + r16) * ldb + koff;

  v8f acc[2][4];
#pragma unroll
  for (int i = 0; i < 2; ++i)
#pragma unroll
    for (int j = 0; j < 4; ++j) acc[i][j] = v8f{};

  for (int k = 0; k < K; k += 32) {
    __builtin_prefetch(pa0 + k + 128, 0, 1);   // global_prefetch_b8
    __builtin_prefetch(pb0 + k + 128, 0, 1);
    bf16x16 a0 = ldfrag(pa0 + k);
    bf16x16 a1 = ldfrag(pa1 + k);
    bf16x16 b0 = ldfrag(pb0 + k);
    bf16x16 b1 = ldfrag(pb1 + k);
    bf16x16 b2 = ldfrag(pb2 + k);
    bf16x16 b3 = ldfrag(pb3 + k);
    acc[0][0] = __builtin_amdgcn_wmma_f32_16x16x32_bf16(false, a0, false, b0, (short)0, acc[0][0], false, false);
    acc[0][1] = __builtin_amdgcn_wmma_f32_16x16x32_bf16(false, a0, false, b1, (short)0, acc[0][1], false, false);
    acc[0][2] = __builtin_amdgcn_wmma_f32_16x16x32_bf16(false, a0, false, b2, (short)0, acc[0][2], false, false);
    acc[0][3] = __builtin_amdgcn_wmma_f32_16x16x32_bf16(false, a0, false, b3, (short)0, acc[0][3], false, false);
    acc[1][0] = __builtin_amdgcn_wmma_f32_16x16x32_bf16(false, a1, false, b0, (short)0, acc[1][0], false, false);
    acc[1][1] = __builtin_amdgcn_wmma_f32_16x16x32_bf16(false, a1, false, b1, (short)0, acc[1][1], false, false);
    acc[1][2] = __builtin_amdgcn_wmma_f32_16x16x32_bf16(false, a1, false, b2, (short)0, acc[1][2], false, false);
    acc[1][3] = __builtin_amdgcn_wmma_f32_16x16x32_bf16(false, a1, false, b3, (short)0, acc[1][3], false, false);
  }

  // C/D layout: VGPR v, lane L -> row = v + 8*(L>>4), col = L&15
  const int rb = (lane >> 4) << 3;
  const int cl = lane & 15;
#pragma unroll
  for (int mi = 0; mi < 2; ++mi) {
#pragma unroll
    for (int ni = 0; ni < 4; ++ni) {
      int mrow0 = m0 + (mi << 4) + rb;
      int col   = n0 + (ni << 4) + cl;
      if (mode == 0) {
#pragma unroll
        for (int v = 0; v < 8; ++v)
          outB[(size_t)(mrow0 + v) * ldc + col] = (__bf16)acc[mi][ni][v];
      } else {
        if (col < n_valid) {
#pragma unroll
          for (int v = 0; v < 8; ++v) {
            int row = mrow0 + v;
            outF[(size_t)(row >> 4) * (PRED_TOT * 16) +
                 (size_t)(p_goff + col) * 16 + (row & 15)] = acc[mi][ni][v];
          }
        }
      }
    }
  }
}

// ---------------------------------------------------------------------------
extern "C" void kernel_launch(void* const* d_in, const int* in_sizes, int n_in,
                              void* d_out, int out_size, void* d_ws, size_t ws_size,
                              hipStream_t stream) {
  (void)in_sizes; (void)n_in; (void)out_size; (void)ws_size;
  const float* tensor = (const float*)d_in[0];
  const float* Wins[3] = {(const float*)d_in[1], (const float*)d_in[2],
                          (const float*)d_in[3]};
  float* out = (float*)d_out;

  char* w = (char*)d_ws;
  auto carve = [&](size_t bytes) {
    char* p = w; w += (bytes + 255) & ~(size_t)255; return p;
  };
  float*  AUG = (float*) carve((size_t)256 * 512 * sizeof(float));
  float*  Ad  = (float*) carve((size_t)256 * 256 * sizeof(float));
  float*  Bd  = (float*) carve((size_t)256 * sizeof(float));
  float*  Km  = (float*) carve((size_t)384 * 256 * sizeof(float));
  float*  G   = (float*) carve((size_t)384 * 32 * sizeof(float));
  __bf16* MT  = (__bf16*)carve((size_t)8192 * 384 * sizeof(__bf16));
  __bf16* Ebf = (__bf16*)carve((size_t)384 * 256 * sizeof(__bf16));
  __bf16* U   = (__bf16*)carve((size_t)1024 * 384 * sizeof(__bf16));
  __bf16* R   = (__bf16*)carve((size_t)1024 * 8192 * sizeof(__bf16));
  __bf16* Wt  = (__bf16*)carve((size_t)256 * 8192 * sizeof(__bf16));
  __bf16* Yb  = (__bf16*)carve((size_t)1024 * 256 * sizeof(__bf16));

  const int wins[3] = {96, 192, 384};
  int p_goff = 0;
  for (int widx = 0; widx < 3; ++widx) {
    const int ws = wins[widx];
    const float dt = 1.0f / (float)ws;
    const int npad = ((ws + 127) / 128) * 128;   // 128, 256, 384

    // -- input-independent constants (cheap, rebuilt deterministically) --
    k_build_aug   <<<(256 * 512 + 255) / 256, 256, 0, stream>>>(AUG, dt);
    k_gauss_jordan<<<1, 256, 0, stream>>>(AUG);
    k_compute_Ad  <<<(256 * 256 + 255) / 256, 256, 0, stream>>>(AUG, Ad, dt);
    k_compute_Bd  <<<1, 256, 0, stream>>>(AUG, Bd, dt);
    k_scan_K      <<<1, 256, 0, stream>>>(Ad, Bd, Km, ws);
    k_gen_G       <<<(ws * 32 + 255) / 256, 256, 0, stream>>>(G, ws);
    {
      long tot = (long)8192 * ws;
      k_gen_MT<<<(int)((tot + 255) / 256), 256, 0, stream>>>(Km, G, MT, ws);
    }
    k_gen_E<<<(npad + 63) / 64, 64, 0, stream>>>(Ebf, ws, npad);

    // -- data path --
    {
      long tot = (long)NROWS * ws;
      k_gen_U<<<(int)((tot + 255) / 256), 256, 0, stream>>>(tensor, U, ws);
    }
    k_gen_Wt<<<(256 * 8192 + 255) / 256, 256, 0, stream>>>(Wins[widx], Wt);

    // GEMM1: R(1024x8192) = U(1024xws) @ M(wsx8192)
    dim3 g1(8192 / 128, NROWS / 128);
    k_wmma_gemm<<<g1, 256, 0, stream>>>(U, ws, MT, ws, ws, 0, R, 8192,
                                        nullptr, 0, 0);
    // GEMM2: Y(1024x256) = R(1024x8192) @ Wbig(8192x256)
    dim3 g2(256 / 128, NROWS / 128);
    k_wmma_gemm<<<g2, 256, 0, stream>>>(R, 8192, Wt, 8192, 8192, 0, Yb, 256,
                                        nullptr, 0, 0);
    // GEMM3: out(1024xws) = Y(1024x256) @ E^T, scattered into d_out
    dim3 g3(npad / 128, NROWS / 128);
    k_wmma_gemm<<<g3, 256, 0, stream>>>(Yb, 256, Ebf, 256, 256, 1, nullptr, 0,
                                        out, ws, p_goff);
    p_goff += ws;
  }
}
